// HookedMambaBlock_4664334483709
// MI455X (gfx1250) — compile-verified, matrix operand order checked by reading, not run on hardware
//
#include <hip/hip_runtime.h>
#include <hip/hip_bf16.h>

// ---------------------------------------------------------------------------
// Mamba block for MI455X (gfx1250, wave32).
// GEMMs: v_wmma_f32_16x16x32_bf16, 32x64 tile per wave (MT=2 x NT=4),
// single-buffered fragments (compiler software-pipelines the loads; explicit
// double-buffering caused scratch spills). Weights pre-packed to
// WMMA-fragment-major bf16. Scan: one lane per (e,n).
// ---------------------------------------------------------------------------

typedef __attribute__((ext_vector_type(16))) __bf16 v16bf;
typedef __attribute__((ext_vector_type(8)))  __bf16 v8bf;
typedef __attribute__((ext_vector_type(8)))  float  v8f;

#define RMS_EPS 1e-5f

static __device__ __forceinline__ float sigmoidf_(float x) {
  return 1.0f / (1.0f + __expf(-x));
}

// ---------------- RMSNorm -> bf16 ------------------------------------------
__global__ void k_rmsnorm(const float* __restrict__ resid,
                          const float* __restrict__ w,
                          __bf16* __restrict__ out, int D) {
  const int l = blockIdx.x;
  const float* row = resid + (size_t)l * D;
  __shared__ float red[256];
  float ss = 0.f;
  for (int d = threadIdx.x; d < D; d += 256) { float v = row[d]; ss += v * v; }
  red[threadIdx.x] = ss;
  __syncthreads();
  for (int s = 128; s > 0; s >>= 1) {
    if (threadIdx.x < s) red[threadIdx.x] += red[threadIdx.x + s];
    __syncthreads();
  }
  const float scale = rsqrtf(red[0] / (float)D + RMS_EPS);
  for (int d = threadIdx.x; d < D; d += 256)
    out[(size_t)l * D + d] = (__bf16)(row[d] * scale * w[d]);
}

// ---------------- Pack fp32 weight [K,N] -> fragment-major bf16 -------------
// Tile (kt,nt) covers K=kt*32..+31, N=nt*16..+15; 512 bf16 per tile stored in
// exact (lane, element) order of the 16x16x32 bf16 B fragment:
//   lane: n = nt*16 + (lane&15)
//   elem e: k = kt*32 + (e&7) + ((e>>3)<<4) + ((lane>>4)<<3)
__global__ void k_pack_b(const float* __restrict__ W, __bf16* __restrict__ out,
                         int K, int N) {
  size_t idx = (size_t)blockIdx.x * blockDim.x + threadIdx.x;
  size_t total = (size_t)K * N;
  if (idx >= total) return;
  size_t tile = idx >> 9;
  int r    = (int)(idx & 511);
  int lane = r >> 4;
  int e    = r & 15;
  int ntiles = N >> 4;
  int kt = (int)(tile / ntiles);
  int nt = (int)(tile % ntiles);
  int k = kt * 32 + (e & 7) + ((e >> 3) << 4) + ((lane >> 4) << 3);
  int n = nt * 16 + (lane & 15);
  out[idx] = (__bf16)W[(size_t)k * N + n];
}

// ---------------- WMMA GEMM: C[M,N] = A[M,K] @ B[K,N] (+ epilogue) ----------
enum { EPI_NONE = 0, EPI_SILU = 1, EPI_BF16 = 2, EPI_SOFTPLUS_BIAS = 3, EPI_ADD = 4 };

template <int MT, int NT, int EPI>
__global__ void k_gemm(const __bf16* __restrict__ A,   // [M,K] row-major bf16
                       const __bf16* __restrict__ Bp,  // packed fragment-major
                       float* __restrict__ C,
                       int M, int N, int K,
                       const float* __restrict__ bias,  // [N] (softplus epi)
                       const float* __restrict__ addm,  // [M,N] (residual epi)
                       __bf16* __restrict__ outbf) {    // [M,N] (bf16 epi)
  const int lane  = threadIdx.x & 31;
  const int wave  = threadIdx.x >> 5;
  const int m0    = (blockIdx.y * 8 + wave) * (16 * MT);
  const int n0    = blockIdx.x * (16 * NT);
  if (m0 >= M) return;
  const int laneM = lane & 15;
  const int half  = lane >> 4;

  v8f acc[MT][NT];
#pragma unroll
  for (int mt = 0; mt < MT; ++mt)
#pragma unroll
    for (int t = 0; t < NT; ++t)
#pragma unroll
      for (int v = 0; v < 8; ++v) acc[mt][t][v] = 0.f;

  const int ntiles = N >> 4;
  const __bf16* arow[MT];
#pragma unroll
  for (int mt = 0; mt < MT; ++mt)
    arow[mt] = A + (size_t)(m0 + mt * 16 + laneM) * K + half * 8;

  for (int k0 = 0; k0 < K; k0 += 32) {
    v16bf afrag[MT];
#pragma unroll
    for (int mt = 0; mt < MT; ++mt) {
      v8bf lo = *(const v8bf*)(arow[mt] + k0);       // K = base .. base+7
      v8bf hi = *(const v8bf*)(arow[mt] + k0 + 16);  // K = base+16 .. base+23
      afrag[mt] = __builtin_shufflevector(lo, hi, 0, 1, 2, 3, 4, 5, 6, 7,
                                          8, 9, 10, 11, 12, 13, 14, 15);
    }
    const __bf16* btile =
        Bp + ((size_t)(k0 >> 5) * ntiles + (n0 >> 4)) * 512 + lane * 16;
#pragma unroll
    for (int t = 0; t < NT; ++t) {
      v16bf bfrag = *(const v16bf*)(btile + (size_t)t * 512);
#pragma unroll
      for (int mt = 0; mt < MT; ++mt)
        acc[mt][t] = __builtin_amdgcn_wmma_f32_16x16x32_bf16(
            false, afrag[mt], false, bfrag, (short)0, acc[mt][t], false,
            false);
    }
  }

#pragma unroll
  for (int mt = 0; mt < MT; ++mt) {
#pragma unroll
    for (int t = 0; t < NT; ++t) {
#pragma unroll
      for (int v = 0; v < 8; ++v) {
        int r = m0 + mt * 16 + v + 8 * half;
        int c = n0 + t * 16 + laneM;
        float val = acc[mt][t][v];
        size_t o = (size_t)r * N + c;
        if (EPI == EPI_NONE) C[o] = val;
        if (EPI == EPI_SILU) C[o] = val * sigmoidf_(val);
        if (EPI == EPI_BF16) outbf[o] = (__bf16)val;
        if (EPI == EPI_SOFTPLUS_BIAS) {
          float z = val + bias[c];
          C[o] = (z > 20.f) ? z : log1pf(__expf(z));
        }
        if (EPI == EPI_ADD) C[o] = val + addm[o];
      }
    }
  }
}

// ---------------- Depthwise causal conv (k=4) + SiLU ------------------------
__global__ void k_conv_silu(const float* __restrict__ xin,  // [L,E]
                            const float* __restrict__ cw,   // [4,1,E]
                            const float* __restrict__ cb,   // [E]
                            float* __restrict__ x,          // [L,E] fp32
                            __bf16* __restrict__ xbf,       // [L,E] bf16
                            int L, int E) {
  size_t idx = (size_t)blockIdx.x * blockDim.x + threadIdx.x;
  if (idx >= (size_t)L * E) return;
  int l = (int)(idx / E);
  int e = (int)(idx % E);
  float acc = cb[e];
#pragma unroll
  for (int k = 0; k < 4; ++k) {
    int ll = l - 3 + k;
    if (ll >= 0) acc += xin[(size_t)ll * E + e] * cw[k * E + e];
  }
  float s = acc * sigmoidf_(acc);
  x[idx] = s;
  xbf[idx] = (__bf16)s;
}

// ---------------- Selective scan: one lane per (e,n) ------------------------
// h[e,n] carried in a register across L steps; y = sum_n h*C via shfl_xor
// inside each 16-lane group. Fuses +x*W_D, *silu(skip), bf16 store of y_skip.
__global__ void k_scan(const float* __restrict__ delta,  // [L,E]
                       const float* __restrict__ Bm,     // [L,16]
                       const float* __restrict__ Cm,     // [L,16]
                       const float* __restrict__ x,      // [L,E]
                       const float* __restrict__ Alog,   // [E,16]
                       const float* __restrict__ WD,     // [E]
                       const float* __restrict__ sskip,  // [L,E] = silu(skip)
                       __bf16* __restrict__ yskip_bf,    // [L,E]
                       int L, int E) {
  int t = blockIdx.x * blockDim.x + threadIdx.x;  // E*16 threads
  int n = t & 15;
  int e = t >> 4;
  if (e >= E) return;
  const float a  = -__expf(Alog[e * 16 + n]);
  const float wd = WD[e];
  float h = 0.f;
  for (int l = 0; l < L; ++l) {
    float d  = delta[(size_t)l * E + e];
    float b  = Bm[l * 16 + n];
    float c  = Cm[l * 16 + n];
    float xv = x[(size_t)l * E + e];
    h = h * __expf(d * a) + (d * xv) * b;
    float y = h * c;
    // reduce over the 16 n-lanes (wave32: xor masks < 16 stay in-group)
    y += __shfl_xor(y, 1, 32);
    y += __shfl_xor(y, 2, 32);
    y += __shfl_xor(y, 4, 32);
    y += __shfl_xor(y, 8, 32);
    if (n == 0) {
      float ys = (y + xv * wd) * sskip[(size_t)l * E + e];
      yskip_bf[(size_t)l * E + e] = (__bf16)ys;
    }
  }
}

// ---------------------------------------------------------------------------
extern "C" void kernel_launch(void* const* d_in, const int* in_sizes, int n_in,
                              void* d_out, int out_size, void* d_ws,
                              size_t ws_size, hipStream_t stream) {
  (void)in_sizes; (void)n_in; (void)out_size; (void)ws_size;
  constexpr int L = 2048, D = 2048, E = 4096, DD = 128;

  const float* resid   = (const float*)d_in[0];
  const float* norm_w  = (const float*)d_in[1];
  const float* W_skip  = (const float*)d_in[2];
  const float* W_in    = (const float*)d_in[3];
  const float* conv_w  = (const float*)d_in[4];
  const float* conv_b  = (const float*)d_in[5];
  const float* W_d1    = (const float*)d_in[6];
  const float* W_d2    = (const float*)d_in[7];
  const float* b_d2    = (const float*)d_in[8];
  const float* W_B     = (const float*)d_in[9];
  const float* W_C     = (const float*)d_in[10];
  const float* A_log   = (const float*)d_in[11];
  const float* W_D     = (const float*)d_in[12];
  const float* W_out   = (const float*)d_in[13];

  char* ws = (char*)d_ws;
  size_t off = 0;
  auto alloc = [&](size_t bytes) -> char* {
    char* p = ws + off;
    off = (off + bytes + 255) & ~(size_t)255;
    return p;
  };
  __bf16* normbf  = (__bf16*)alloc((size_t)L * D * 2);
  __bf16* Wskip_p = (__bf16*)alloc((size_t)D * E * 2);
  __bf16* Win_p   = (__bf16*)alloc((size_t)D * E * 2);
  __bf16* Wd1_p   = (__bf16*)alloc((size_t)E * DD * 2);
  __bf16* Wd2_p   = (__bf16*)alloc((size_t)DD * E * 2);
  __bf16* WB_p    = (__bf16*)alloc((size_t)E * 16 * 2);
  __bf16* WC_p    = (__bf16*)alloc((size_t)E * 16 * 2);
  __bf16* Wout_p  = (__bf16*)alloc((size_t)E * D * 2);
  float*  sskip   = (float*)alloc((size_t)L * E * 4);
  float*  xin     = (float*)alloc((size_t)L * E * 4);  // reused as delta
  float*  xf      = (float*)alloc((size_t)L * E * 4);
  __bf16* xbf     = (__bf16*)alloc((size_t)L * E * 2);
  __bf16* dt1bf   = (__bf16*)alloc((size_t)L * DD * 2);
  float*  Bm      = (float*)alloc((size_t)L * 16 * 4);
  float*  Cm      = (float*)alloc((size_t)L * 16 * 4);
  __bf16* yskipbf = (__bf16*)alloc((size_t)L * E * 2);
  float*  delta   = xin;  // alias: x_in is dead after the conv kernel

  auto packGrid = [](size_t elems) { return dim3((unsigned)((elems + 255) / 256)); };

  // 1) RMSNorm -> bf16
  k_rmsnorm<<<L, 256, 0, stream>>>(resid, norm_w, normbf, D);

  // 2) Pack weights to WMMA fragment-major bf16
  k_pack_b<<<packGrid((size_t)D * E), 256, 0, stream>>>(W_skip, Wskip_p, D, E);
  k_pack_b<<<packGrid((size_t)D * E), 256, 0, stream>>>(W_in, Win_p, D, E);
  k_pack_b<<<packGrid((size_t)E * DD), 256, 0, stream>>>(W_d1, Wd1_p, E, DD);
  k_pack_b<<<packGrid((size_t)DD * E), 256, 0, stream>>>(W_d2, Wd2_p, DD, E);
  k_pack_b<<<packGrid((size_t)E * 16), 256, 0, stream>>>(W_B, WB_p, E, 16);
  k_pack_b<<<packGrid((size_t)E * 16), 256, 0, stream>>>(W_C, WC_p, E, 16);
  k_pack_b<<<packGrid((size_t)E * D), 256, 0, stream>>>(W_out, Wout_p, E, D);

  // Each block: 8 waves x (32 rows x 64 cols) = 256 rows x 64 cols
  // 3) sskip = silu(norm @ W_skip)    [L,E]
  k_gemm<2, 4, EPI_SILU><<<dim3(E / 64, L / 256), 256, 0, stream>>>(
      normbf, Wskip_p, sskip, L, E, D, nullptr, nullptr, nullptr);

  // 4) x_in = norm @ W_in             [L,E]
  k_gemm<2, 4, EPI_NONE><<<dim3(E / 64, L / 256), 256, 0, stream>>>(
      normbf, Win_p, xin, L, E, D, nullptr, nullptr, nullptr);

  // 5) x = silu(causal_conv(x_in))    [L,E] fp32 + bf16
  k_conv_silu<<<packGrid((size_t)L * E), 256, 0, stream>>>(
      xin, conv_w, conv_b, xf, xbf, L, E);

  // 6) dt1 = x @ W_delta_1            [L,128] bf16
  k_gemm<2, 4, EPI_BF16><<<dim3(DD / 64, L / 256), 256, 0, stream>>>(
      xbf, Wd1_p, nullptr, L, DD, E, nullptr, nullptr, dt1bf);

  // 7) delta = softplus(dt1 @ W_delta_2 + b)   [L,E]  (aliases xin buffer)
  k_gemm<2, 4, EPI_SOFTPLUS_BIAS><<<dim3(E / 64, L / 256), 256, 0, stream>>>(
      dt1bf, Wd2_p, delta, L, E, DD, b_d2, nullptr, nullptr);

  // 8) Bm = x @ W_B, Cm = x @ W_C     [L,16]
  k_gemm<2, 1, EPI_NONE><<<dim3(1, L / 256), 256, 0, stream>>>(
      xbf, WB_p, Bm, L, 16, E, nullptr, nullptr, nullptr);
  k_gemm<2, 1, EPI_NONE><<<dim3(1, L / 256), 256, 0, stream>>>(
      xbf, WC_p, Cm, L, 16, E, nullptr, nullptr, nullptr);

  // 9) selective scan + D path + silu(skip) gate -> y_skip bf16 [L,E]
  k_scan<<<(E * 16) / 256, 256, 0, stream>>>(delta, Bm, Cm, xf, A_log, W_D,
                                             sskip, yskipbf, L, E);

  // 10) out = resid + y_skip @ W_out  [L,D]
  k_gemm<2, 4, EPI_ADD><<<dim3(D / 64, L / 256), 256, 0, stream>>>(
      yskipbf, Wout_p, (float*)d_out, L, D, E, nullptr, resid, nullptr);
}
